// MyLstmModel_32014686225102
// MI455X (gfx1250) — matrix operand order repaired and checked by reading, not compiled, not run on hardware
//
#include <hip/hip_runtime.h>
#include <hip/hip_bf16.h>

typedef __attribute__((ext_vector_type(16))) _Float16 v16h;
typedef __attribute__((ext_vector_type(4)))  _Float16 v4h;
typedef __attribute__((ext_vector_type(2)))  _Float16 v2h;
typedef __attribute__((ext_vector_type(8)))  float    v8f;

#define B_TOT 4096
#define T_LEN 128
#define F_IN  64
#define H1    32
#define H2    16

// Branch-free fast activations: v_exp_f32 + v_rcp_f32, no divergence.
__device__ __forceinline__ float sigf(float x) {
    return __builtin_amdgcn_rcpf(1.0f + __expf(-x));
}
__device__ __forceinline__ float tanhf_fast(float x) {
    // tanh(x) = 1 - 2/(1+e^{2x});  saturates correctly at +/-1 for |x| large
    return 1.0f - 2.0f * __builtin_amdgcn_rcpf(1.0f + __expf(2.0f * x));
}

// ---------------------------------------------------------------------------
// Layer 1: bidirectional LSTM, H1=32, return_sequences -> h1 [B,T,64] (f16)
// grid = (B/16, 2 directions), block = 256 (8 waves).
// Wave w computes gate columns [16w, 16w+16) of z = x_t@W + h@U + b.
// W/U slices live in registers as WMMA B-fragments across the whole scan;
// the x_t tile load is software-pipelined one timestep ahead in registers and
// published to LDS already converted to f16 (fragment build = pure ds loads).
// ---------------------------------------------------------------------------
__global__ __launch_bounds__(256)
void lstm1_kernel(const float* __restrict__ x,
                  const float* __restrict__ Wf, const float* __restrict__ Uf,
                  const float* __restrict__ bf,
                  const float* __restrict__ Wb, const float* __restrict__ Ub,
                  const float* __restrict__ bb,
                  _Float16* __restrict__ h1out)
{
    const int dir  = blockIdx.y;
    const int b0   = blockIdx.x * 16;
    const int tid  = threadIdx.x;
    const int wave = tid >> 5;
    const int lane = tid & 31;
    const int n    = lane & 15;   // column within tile / A-row
    const int half = lane >> 4;

    const float* W    = dir ? Wb : Wf;
    const float* U    = dir ? Ub : Uf;
    const float* bias = dir ? bb : bf;

    __shared__ _Float16 xs[16][72];    // x_t tile [16 x 64] f16 (144B rows)
    __shared__ float    zs[16][132];   // z tile [16 x 128] f32 (padded)
    __shared__ float    cs[16][32];    // cell state
    __shared__ _Float16 hs[16][36];    // hidden state (f16, padded; 72B rows)

    // --- time-invariant B fragments (contiguous K order, same as A: lanes 0-15 K=0-15,
    // lanes 16-31 K=16-31, VGPR v packs K = half*16 + 2v, +1) ---
    v16h bW0, bW1, bU;
    const int col = wave * 16 + n;     // gate column 0..127
    #pragma unroll
    for (int v = 0; v < 8; ++v) {
        const int k0 = half * 16 + 2 * v;
        bW0[2*v]   = (_Float16)W[(k0     ) * (4*H1) + col];
        bW0[2*v+1] = (_Float16)W[(k0 +  1) * (4*H1) + col];
        bW1[2*v]   = (_Float16)W[(k0 + 32) * (4*H1) + col];
        bW1[2*v+1] = (_Float16)W[(k0 + 33) * (4*H1) + col];
        bU [2*v]   = (_Float16)U[(k0     ) * (4*H1) + col];
        bU [2*v+1] = (_Float16)U[(k0 +  1) * (4*H1) + col];
    }
    const float biasv = bias[col];     // folded into accumulator init

    // --- zero initial state (each thread owns one (m, j-pair)) ---
    {
        const int m = tid >> 4;
        const int j = (tid & 15) * 2;
        cs[m][j] = 0.0f; cs[m][j+1] = 0.0f;
        v2h z2; z2[0] = (_Float16)0.0f; z2[1] = (_Float16)0.0f;
        *(v2h*)(&hs[m][j]) = z2;
    }
    __syncthreads();

    // --- pipelined x loader state (converted to f16 at publish time) ---
    const int lm = tid >> 4;
    const int lf = (tid & 15) * 4;
    const float* xbase = x + (size_t)(b0 + lm) * T_LEN * F_IN + lf;
    float4 xreg = *(const float4*)(xbase + (size_t)(dir ? (T_LEN - 1) : 0) * F_IN);

    for (int t = 0; t < T_LEN; ++t) {
        const int tt = dir ? (T_LEN - 1 - t) : t;

        // publish prefetched x_t tile as f16, then issue next load
        {
            v4h xh;
            xh[0] = (_Float16)xreg.x; xh[1] = (_Float16)xreg.y;
            xh[2] = (_Float16)xreg.z; xh[3] = (_Float16)xreg.w;
            *(v4h*)(&xs[lm][lf]) = xh;    // 8B store, aligned (144B rows)
        }
        __syncthreads();
        if (t + 1 < T_LEN) {
            const int tn = dir ? (tt - 1) : (tt + 1);
            xreg = *(const float4*)(xbase + (size_t)tn * F_IN);
        }

        // --- build A fragments (lane row = n,
        // VGPR v -> K = half*16 + 2*v, +1; matches B) ---
        v16h ax0, ax1, ah;
        #pragma unroll
        for (int v = 0; v < 8; ++v) {
            const int k = half * 16 + 2 * v;
            ax0[2*v]   = xs[n][k];
            ax0[2*v+1] = xs[n][k + 1];
            ax1[2*v]   = xs[n][k + 32];
            ax1[2*v+1] = xs[n][k + 33];
            ah [2*v]   = hs[n][k];
            ah [2*v+1] = hs[n][k + 1];
        }

        v8f acc;
        #pragma unroll
        for (int r = 0; r < 8; ++r) acc[r] = biasv;   // bias pre-folded
        acc = __builtin_amdgcn_wmma_f32_16x16x32_f16(false, ax0, false, bW0,
                                                     (short)0, acc, false, false);
        acc = __builtin_amdgcn_wmma_f32_16x16x32_f16(false, ax1, false, bW1,
                                                     (short)0, acc, false, false);
        acc = __builtin_amdgcn_wmma_f32_16x16x32_f16(false, ah,  false, bU,
                                                     (short)0, acc, false, false);

        // scatter D tile to LDS (D layout: VGPR r -> row r + half*8, col n)
        #pragma unroll
        for (int r = 0; r < 8; ++r)
            zs[r + half * 8][col] = acc[r];
        __syncthreads();

        // --- gates + state update: thread owns (m, j) and (m, j+1) ---
        {
            const int m = tid >> 4;          // 0..15
            const int j = (tid & 15) * 2;    // 0,2,..,30
            float hv[2];
            v2h hh;
            #pragma unroll
            for (int q = 0; q < 2; ++q) {
                const int jq = j + q;
                const float zi = zs[m][jq         ];
                const float zf = zs[m][jq +     H1];
                const float zg = zs[m][jq + 2 * H1];
                const float zo = zs[m][jq + 3 * H1];
                const float ig = sigf(zi);
                const float fg = sigf(zf);
                const float gg = tanhf_fast(zg);
                const float og = sigf(zo);
                const float c  = fg * cs[m][jq] + ig * gg;
                cs[m][jq] = c;
                hv[q] = og * tanhf_fast(c);
                hh[q] = (_Float16)hv[q];
            }
            *(v2h*)(&hs[m][j]) = hh;   // row stride 72B, j even -> 4B aligned
            // fwd -> cols [0,32), bwd -> cols [32,64) at un-reversed time tt
            *(v2h*)(h1out + ((size_t)(b0 + m) * T_LEN + tt) * (2 * H1)
                          + dir * H1 + j) = hh;
        }
        __syncthreads();
    }
}

// ---------------------------------------------------------------------------
// Layer 2: bidirectional LSTM, H2=16, last hidden only -> h2cat [B,32] (f32)
// grid = (B/16, 2), block = 128 (4 waves). Wave w owns gate cols [16w,16w+16).
// ---------------------------------------------------------------------------
__global__ __launch_bounds__(128)
void lstm2_kernel(const _Float16* __restrict__ h1,
                  const float* __restrict__ Wf, const float* __restrict__ Uf,
                  const float* __restrict__ bf,
                  const float* __restrict__ Wb, const float* __restrict__ Ub,
                  const float* __restrict__ bb,
                  float* __restrict__ h2cat)
{
    const int dir  = blockIdx.y;
    const int b0   = blockIdx.x * 16;
    const int tid  = threadIdx.x;
    const int wave = tid >> 5;
    const int lane = tid & 31;
    const int n    = lane & 15;
    const int half = lane >> 4;

    const float* W    = dir ? Wb : Wf;
    const float* U    = dir ? Ub : Uf;
    const float* bias = dir ? bb : bf;

    __shared__ _Float16 h1s[16][72];   // input tile [16 x 64] f16 (padded)
    __shared__ float    zs[16][68];    // z tile [16 x 64] f32 (padded)
    __shared__ float    cs[16][16];
    __shared__ _Float16 hs[16][20];    // 40B rows

    v16h bW0, bW1, bU;
    const int col = wave * 16 + n;     // gate column 0..63
    #pragma unroll
    for (int v = 0; v < 8; ++v) {
        const int k0 = half * 16 + 2 * v;
        bW0[2*v]   = (_Float16)W[(k0     ) * (4*H2) + col];
        bW0[2*v+1] = (_Float16)W[(k0 +  1) * (4*H2) + col];
        bW1[2*v]   = (_Float16)W[(k0 + 32) * (4*H2) + col];
        bW1[2*v+1] = (_Float16)W[(k0 + 33) * (4*H2) + col];
        // U2 has only K=16 real rows; zero-pad K >= 16 (half==1 lanes)
        if (half == 0) {
            bU[2*v]   = (_Float16)U[(k0    ) * (4*H2) + col];
            bU[2*v+1] = (_Float16)U[(k0 + 1) * (4*H2) + col];
        } else {
            bU[2*v]   = (_Float16)0.0f;
            bU[2*v+1] = (_Float16)0.0f;
        }
    }
    const float biasv = bias[col];

    {
        const int m = tid >> 3;
        const int j = (tid & 7) * 2;
        cs[m][j] = 0.0f; cs[m][j+1] = 0.0f;
        v2h z2; z2[0] = (_Float16)0.0f; z2[1] = (_Float16)0.0f;
        *(v2h*)(&hs[m][j]) = z2;
    }
    __syncthreads();

    // --- pipelined h1-tile loader: 128 threads x 8 f16 (16B) ---
    const int lm = tid >> 3;
    const int lf = (tid & 7) * 8;
    const _Float16* h1base = h1 + (size_t)(b0 + lm) * T_LEN * (2 * H1) + lf;
    float4 hreg = *(const float4*)(h1base + (size_t)(dir ? (T_LEN - 1) : 0) * (2 * H1));

    for (int t = 0; t < T_LEN; ++t) {
        const int tt = dir ? (T_LEN - 1 - t) : t;

        *(float4*)(&h1s[lm][lf]) = hreg;
        __syncthreads();
        if (t + 1 < T_LEN) {
            const int tn = dir ? (tt - 1) : (tt + 1);
            hreg = *(const float4*)(h1base + (size_t)tn * (2 * H1));
        }

        v16h ax0, ax1, ah;
        #pragma unroll
        for (int v = 0; v < 8; ++v) {
            const int g = v >> 2;
            const int k = half * 16 + 2 * v;
            ax0[2*v]   = h1s[n][k];
            ax0[2*v+1] = h1s[n][k + 1];
            ax1[2*v]   = h1s[n][k + 32];
            ax1[2*v+1] = h1s[n][k + 33];
            if (half == 0) {       // real K = 0..15 (recurrent width 16)
                ah[2*v]   = hs[n][2 * v];
                ah[2*v+1] = hs[n][2 * v + 1];
            } else {               // zero-pad K = 16..31
                ah[2*v]   = (_Float16)0.0f;
                ah[2*v+1] = (_Float16)0.0f;
            }
        }

        v8f acc;
        #pragma unroll
        for (int r = 0; r < 8; ++r) acc[r] = biasv;
        acc = __builtin_amdgcn_wmma_f32_16x16x32_f16(false, ax0, false, bW0,
                                                     (short)0, acc, false, false);
        acc = __builtin_amdgcn_wmma_f32_16x16x32_f16(false, ax1, false, bW1,
                                                     (short)0, acc, false, false);
        acc = __builtin_amdgcn_wmma_f32_16x16x32_f16(false, ah,  false, bU,
                                                     (short)0, acc, false, false);

        #pragma unroll
        for (int r = 0; r < 8; ++r)
            zs[r + half * 8][col] = acc[r];
        __syncthreads();

        {
            const int m = tid >> 3;          // 0..15
            const int j = (tid & 7) * 2;     // 0,2,..,14
            v2h hh;
            float hv[2];
            #pragma unroll
            for (int q = 0; q < 2; ++q) {
                const int jq = j + q;
                const float zi = zs[m][jq         ];
                const float zf = zs[m][jq +     H2];
                const float zg = zs[m][jq + 2 * H2];
                const float zo = zs[m][jq + 3 * H2];
                const float ig = sigf(zi);
                const float fg = sigf(zf);
                const float gg = tanhf_fast(zg);
                const float og = sigf(zo);
                const float c  = fg * cs[m][jq] + ig * gg;
                cs[m][jq] = c;
                hv[q] = og * tanhf_fast(c);
                hh[q] = (_Float16)hv[q];
            }
            *(v2h*)(&hs[m][j]) = hh;
            if (t == T_LEN - 1) {  // last hidden only (return_sequences=False)
                float* dst = h2cat + (size_t)(b0 + m) * (2 * H2) + dir * H2 + j;
                dst[0] = hv[0];
                dst[1] = hv[1];
            }
        }
        __syncthreads();
    }
}

// ---------------------------------------------------------------------------
// Head: [B,32] @ W3[32,8] + b3 -> swish -> @ W4[8,2] + b4 -> sigmoid
// ---------------------------------------------------------------------------
__global__ __launch_bounds__(256)
void head_kernel(const float* __restrict__ h2cat,
                 const float* __restrict__ W3, const float* __restrict__ b3,
                 const float* __restrict__ W4, const float* __restrict__ b4,
                 float* __restrict__ out)
{
    const int row = blockIdx.x * blockDim.x + threadIdx.x;
    if (row >= B_TOT) return;
    const float* xr = h2cat + (size_t)row * (2 * H2);

    float y[8];
    #pragma unroll
    for (int j = 0; j < 8; ++j) {
        float s = b3[j];
        #pragma unroll
        for (int k = 0; k < 2 * H2; ++k) s += xr[k] * W3[k * 8 + j];
        y[j] = s * sigf(s);                    // swish
    }
    #pragma unroll
    for (int j = 0; j < 2; ++j) {
        float s = b4[j];
        #pragma unroll
        for (int k = 0; k < 8; ++k) s += y[k] * W4[k * 2 + j];
        out[(size_t)row * 2 + j] = sigf(s);
    }
}

// ---------------------------------------------------------------------------
extern "C" void kernel_launch(void* const* d_in, const int* in_sizes, int n_in,
                              void* d_out, int out_size, void* d_ws, size_t ws_size,
                              hipStream_t stream)
{
    const float* x   = (const float*)d_in[0];
    const float* W1f = (const float*)d_in[1];
    const float* U1f = (const float*)d_in[2];
    const float* b1f = (const float*)d_in[3];
    const float* W1b = (const float*)d_in[4];
    const float* U1b = (const float*)d_in[5];
    const float* b1b = (const float*)d_in[6];
    const float* W2f = (const float*)d_in[7];
    const float* U2f = (const float*)d_in[8];
    const float* b2f = (const float*)d_in[9];
    const float* W2b = (const float*)d_in[10];
    const float* U2b = (const float*)d_in[11];
    const float* b2b = (const float*)d_in[12];
    const float* W3  = (const float*)d_in[13];
    const float* b3  = (const float*)d_in[14];
    const float* W4  = (const float*)d_in[15];
    const float* b4  = (const float*)d_in[16];

    char* ws = (char*)d_ws;
    _Float16* h1    = (_Float16*)ws;                 // [B,T,64] f16 = 64 MB
    float*    h2cat = (float*)(ws + (size_t)B_TOT * T_LEN * (2 * H1) * sizeof(_Float16));

    dim3 g1(B_TOT / 16, 2);
    lstm1_kernel<<<g1, dim3(256), 0, stream>>>(x, W1f, U1f, b1f, W1b, U1b, b1b, h1);

    dim3 g2(B_TOT / 16, 2);
    lstm2_kernel<<<g2, dim3(128), 0, stream>>>(h1, W2f, U2f, b2f, W2b, U2b, b2b, h2cat);

    head_kernel<<<dim3(B_TOT / 256), dim3(256), 0, stream>>>(h2cat, W3, b3, W4, b4,
                                                             (float*)d_out);
}
